// ULTODecoder2d_84224308675210
// MI455X (gfx1250) — matrix operand order repaired and checked by reading, not compiled, not run on hardware
//
#include <hip/hip_runtime.h>

typedef __bf16 bf16;
typedef __attribute__((ext_vector_type(16))) __bf16 v16bf;
typedef __attribute__((ext_vector_type(8)))  float  v8f;

#define B_    2
#define M_    2048
#define N_    8192
#define TQ    (B_*M_)      // 4096 query tokens
#define TP    (B_*N_)      // 16384 cloud tokens
#define TALL  (TQ+TP)      // 20480 sampled points
#define DIM   384
#define KNN   32
#define NH    12
#define HD    32
#define NBLK  5

// ---------------------------------------------------------------- helpers
__device__ inline float bf2f(unsigned v) {
    union { unsigned u; float f; } w; w.u = v << 16; return w.f;
}

// Async global->LDS 16B copy (ASYNCcnt-tracked; CDNA5 ISA)
__device__ inline void async_copy_b128(void* lds_ptr, const void* gptr) {
    unsigned lds_off = (unsigned)(unsigned long long)lds_ptr; // LDS aperture: addr[31:0]
    asm volatile("global_load_async_to_lds_b128 %0, %1, off"
                 :: "v"(lds_off), "v"(gptr) : "memory");
}
__device__ inline void wait_async() {
    asm volatile("s_wait_asynccnt 0x0" ::: "memory");
}

// load one 16x(k) WMMA fragment (16 bf16, 32B) from LDS via 2x b128
__device__ inline v16bf load_frag(const bf16* p) {
    union { float4 f[2]; v16bf v; } u;
    u.f[0] = *(const float4*)p;
    u.f[1] = *(const float4*)(p + 8);
    return u.v;
}

// ---------------------------------------------------------------- cast f32 -> bf16
__global__ __launch_bounds__(256) void cast_bf16_kernel(const float* __restrict__ s,
                                                        bf16* __restrict__ d, int n) {
    int i = blockIdx.x * 256 + threadIdx.x;
    if (i < n) d[i] = (bf16)s[i];
}

// ---------------------------------------------------------------- weight pack: f32 [K,N] row-major
// -> bf16 in WMMA B-fragment order: tile(kt,nt) = 1KB block of [lane][i] with
//   lane = (n&15) + ((k&16)?16:0), i = k&15   (contiguous k order; A loader uses the same)
__global__ __launch_bounds__(256) void pack_weight_kernel(const float* __restrict__ w,
                                                          bf16* __restrict__ dst, int K, int N) {
    int gid = blockIdx.x * 256 + threadIdx.x;
    if (gid >= K * N) return;
    int k = gid / N, n = gid - k * N;
    int kt = k >> 5, nt = n >> 4;
    int lane = (n & 15) + (((k >> 4) & 1) << 4);
    int i = k & 15;
    dst[((size_t)(kt * (N >> 4)) + nt) * 512 + lane * 16 + i] = (bf16)w[gid];
}

// ---------------------------------------------------------------- WMMA GEMM
// C[M,N] = act(A[M,K](bf16 row-major) @ Bp(packed bf16) + bias + resid)
// block tile 128x64, 8 waves, wave tile 16x64 (4 WMMA / k-step / wave)
// M%128==0, N%64==0, K%32==0
#define LSTR 24   // padded lane stride (bf16) -> 48B, keeps b128 alignment, low bank conflict
__global__ __launch_bounds__(256) void gemm_bf16_kernel(
    const bf16* __restrict__ A, const bf16* __restrict__ Bp,
    const float* __restrict__ bias, const float* __restrict__ resid,
    float* __restrict__ Cf, bf16* __restrict__ Cb,
    int M, int N, int K, int relu)
{
    __shared__ bf16 sA[8][32][LSTR];   // 8 m-tiles, fragment-major
    __shared__ bf16 sB[4][32][LSTR];   // 4 n-tiles, fragment-major

    const int tid  = threadIdx.x;
    const int bm   = blockIdx.x * 128;
    const int bn   = blockIdx.y * 64;
    const int wave = tid >> 5, lane = tid & 31;
    const int grp = lane >> 4, ln = lane & 15;
    const int nt16 = N >> 4;

    // A loader coords: thread covers rows ar, ar+64 at k-octet ac
    const int ar = tid >> 2, ac = (tid & 3) * 8;
    const int a_lane = (ar & 15) + (((ac >> 4) & 1) << 4);
    const int a_i0   = (ac & 8) ? 8 : 0;
    // B loader coords: thread covers one 16B chunk of the 4KB packed tile
    const int b_ct = tid >> 6, b_lane = (tid >> 1) & 31, b_half = (tid & 1) * 8;

    v8f acc[4] = {};

    for (int k0 = 0; k0 < K; k0 += 32) {
        // ---- async fill A (swizzled per-lane dests) and B (contiguous packed copy)
        async_copy_b128(&sA[(ar >> 4)][a_lane][a_i0],
                        A + (size_t)(bm + ar) * K + k0 + ac);
        async_copy_b128(&sA[((ar + 64) >> 4)][a_lane][a_i0],
                        A + (size_t)(bm + ar + 64) * K + k0 + ac);
        const bf16* bsrc = Bp + ((size_t)((k0 >> 5) * nt16) + (bn >> 4)) * 512;
        async_copy_b128(&sB[b_ct][b_lane][b_half],
                        bsrc + (size_t)(b_ct * 32 + b_lane) * 16 + b_half);
        if (k0 + 32 < K) {   // prefetch next A tile -> global_prefetch_b8
            __builtin_prefetch(A + (size_t)(bm + ar) * K + k0 + 32 + ac, 0, 3);
        }
        wait_async();
        __syncthreads();

        // ---- compute: per wave 1 A frag, 4 B frags, 4 WMMA
        v16bf a = load_frag(&sA[wave][lane][0]);
#pragma unroll
        for (int ct = 0; ct < 4; ++ct) {
            v16bf b = load_frag(&sB[ct][lane][0]);
            acc[ct] = __builtin_amdgcn_wmma_f32_16x16x32_bf16(false, a, false, b,
                                                              (short)0, acc[ct], false, false);
        }
        __syncthreads();
    }

#pragma unroll
    for (int ct = 0; ct < 4; ++ct) {
#pragma unroll
        for (int r = 0; r < 8; ++r) {
            const int row = bm + wave * 16 + r + grp * 8;   // D layout (ISA 7.12.2)
            const int col = bn + ct * 16 + ln;
            float v = acc[ct][r];
            if (bias)  v += bias[col];
            if (resid) v += resid[(size_t)row * N + col];
            if (relu)  v = fmaxf(v, 0.f);
            if (Cf) Cf[(size_t)row * N + col] = v;
            if (Cb) Cb[(size_t)row * N + col] = (bf16)v;
        }
    }
}

// ---------------------------------------------------------------- triplane sampling
__device__ inline float bilin(const float* __restrict__ pl, int C, int R, int b, int c,
                              float cu, float cv) {
    float u = cu / 1.101f + 0.5f;  u = fminf(fmaxf(u, 1e-6f), 1.f - 1e-6f);
    float v = cv / 1.101f + 0.5f;  v = fminf(fmaxf(v, 1e-6f), 1.f - 1e-6f);
    float ix = u * (R - 1), iy = v * (R - 1);
    float fx = floorf(ix), fy = floorf(iy);
    float wx = ix - fx, wy = iy - fy;
    int x0 = min(max((int)fx, 0), R - 1), x1 = min(x0 + 1, R - 1);
    int y0 = min(max((int)fy, 0), R - 1), y1 = min(y0 + 1, R - 1);
    const float* base = pl + ((size_t)b * C + c) * R * R;
    float v00 = base[y0 * R + x0], v01 = base[y0 * R + x1];
    float v10 = base[y1 * R + x0], v11 = base[y1 * R + x1];
    return v00 * (1 - wx) * (1 - wy) + v01 * wx * (1 - wy)
         + v10 * (1 - wx) * wy       + v11 * wx * wy;
}

__global__ __launch_bounds__(256) void triplane_sample_kernel(
    const float* __restrict__ pts, const float* __restrict__ pxz,
    const float* __restrict__ pxy, const float* __restrict__ pyz,
    bf16* __restrict__ feat, int C, int R)
{
    int gid = blockIdx.x * 256 + threadIdx.x;
    if (gid >= TALL * C) return;
    int pt = gid / C, c = gid - pt * C;
    int b = (pt < TQ) ? (pt >> 11) : ((pt - TQ) >> 13);
    float x = pts[pt * 3], y = pts[pt * 3 + 1], z = pts[pt * 3 + 2];
    float acc = bilin(pxz, C, R, b, c, x, z)
              + bilin(pxy, C, R, b, c, x, y)
              + bilin(pyz, C, R, b, c, y, z);
    feat[gid] = (bf16)acc;
}

// ---------------------------------------------------------------- c_full concat
__global__ __launch_bounds__(256) void build_cfull_kernel(
    const float* __restrict__ c_point, const float* __restrict__ t192,
    float* __restrict__ cf, bf16* __restrict__ cb)
{
    int gid = blockIdx.x * 256 + threadIdx.x;
    if (gid >= TP * DIM) return;
    int r = gid / DIM, c = gid - r * DIM;
    float v = (c < 192) ? c_point[(size_t)r * 192 + c]
                        : t192[(size_t)(TQ + r) * 192 + (c - 192)];
    cf[gid] = v; cb[gid] = (bf16)v;
}

// ---------------------------------------------------------------- LayerNorm -> bf16
__global__ __launch_bounds__(256) void layernorm_kernel(
    const float* __restrict__ x, const float* __restrict__ g, const float* __restrict__ b,
    bf16* __restrict__ y, int T)
{
    int wave = threadIdx.x >> 5, lane = threadIdx.x & 31;
    int t = blockIdx.x * 8 + wave;
    if (t >= T) return;
    const float* xr = x + (size_t)t * DIM;
    float v[12], s = 0.f;
#pragma unroll
    for (int i = 0; i < 12; ++i) { v[i] = xr[lane + i * 32]; s += v[i]; }
#pragma unroll
    for (int o = 16; o; o >>= 1) s += __shfl_xor(s, o, 32);
    float mean = s * (1.f / DIM), q = 0.f;
#pragma unroll
    for (int i = 0; i < 12; ++i) { float d = v[i] - mean; q += d * d; }
#pragma unroll
    for (int o = 16; o; o >>= 1) q += __shfl_xor(q, o, 32);
    float rs = rsqrtf(q * (1.f / DIM) + 1e-5f);
#pragma unroll
    for (int i = 0; i < 12; ++i) {
        int d = lane + i * 32;
        y[(size_t)t * DIM + d] = (bf16)((v[i] - mean) * rs * g[d] + b[d]);
    }
}

// ---------------------------------------------------------------- RoPE-3D
__global__ __launch_bounds__(256) void rope_kernel(
    const float* __restrict__ x, const float* __restrict__ xyz,
    float* __restrict__ of, bf16* __restrict__ ob, int T)
{
    int gid = blockIdx.x * 256 + threadIdx.x;
    if (gid >= T * 192) return;
    int t = gid / 192, r = gid - t * 192;
    int axis = r >> 6, j = r & 63;
    float f = xyz[t * 3 + axis] * __powf(10000.f, -(float)j * (1.f / 64.f));
    float c, s; __sincosf(f, &s, &c);
    size_t base = (size_t)t * DIM + axis * 128;
    float x1 = x[base + j], x2 = x[base + 64 + j];
    float o1 = x1 * c - x2 * s;
    float o2 = x2 * c + x1 * s;
    if (of) { of[base + j] = o1; of[base + 64 + j] = o2; }
    if (ob) { ob[base + j] = (bf16)o1; ob[base + 64 + j] = (bf16)o2; }
}

// ---------------------------------------------------------------- brute-force kNN (wave32 merge)
__global__ __launch_bounds__(64) void knn_kernel(
    const float* __restrict__ query, const float* __restrict__ p, int* __restrict__ idx)
{
    __shared__ float dl[2][32 * KNN];
    __shared__ int   il[2][32 * KNN];
    int wave = threadIdx.x >> 5, lane = threadIdx.x & 31;
    int t = blockIdx.x * 2 + wave;
    int b = t >> 11;
    float qx = query[t * 3], qy = query[t * 3 + 1], qz = query[t * 3 + 2];
    float bd[KNN]; int bi[KNN];
#pragma unroll
    for (int i = 0; i < KNN; ++i) { bd[i] = 3.4e38f; bi[i] = 0; }
    const float* pb = p + (size_t)b * N_ * 3;
    for (int n = lane; n < N_; n += 32) {
        float dx = qx - pb[n * 3], dy = qy - pb[n * 3 + 1], dz = qz - pb[n * 3 + 2];
        float d = dx * dx + dy * dy + dz * dz;
        if (d < bd[KNN - 1]) {
            bd[KNN - 1] = d; bi[KNN - 1] = n;
#pragma unroll
            for (int j = KNN - 1; j > 0; --j)
                if (bd[j] < bd[j - 1]) {
                    float td = bd[j]; bd[j] = bd[j - 1]; bd[j - 1] = td;
                    int   ti = bi[j]; bi[j] = bi[j - 1]; bi[j - 1] = ti;
                }
        }
    }
#pragma unroll
    for (int i = 0; i < KNN; ++i) { dl[wave][lane * KNN + i] = bd[i]; il[wave][lane * KNN + i] = bi[i]; }
    int head = 0;
    for (int r = 0; r < KNN; ++r) {
        float cand = (head < KNN) ? dl[wave][lane * KNN + head] : 3.4e38f;
        int wl = lane;
#pragma unroll
        for (int o = 16; o; o >>= 1) {
            float oc = __shfl_xor(cand, o, 32);
            int   ol = __shfl_xor(wl, o, 32);
            if (oc < cand || (oc == cand && ol < wl)) { cand = oc; wl = ol; }
        }
        if (lane == wl) { idx[t * KNN + r] = il[wave][lane * KNN + head]; head++; }
    }
}

// ---------------------------------------------------------------- local attention (gathered K/V)
__device__ inline float dot_row(const float* q, const bf16* kp) {
    const uint4* pp = (const uint4*)kp;
    float acc = 0.f;
#pragma unroll
    for (int i = 0; i < 4; ++i) {
        uint4 u = pp[i];
        unsigned w[4] = { u.x, u.y, u.z, u.w };
#pragma unroll
        for (int j = 0; j < 4; ++j) {
            acc += q[i * 8 + j * 2]     * bf2f(w[j] & 0xffffu);
            acc += q[i * 8 + j * 2 + 1] * bf2f(w[j] >> 16);
        }
    }
    return acc;
}
__device__ inline void axpy_row(float w0, const bf16* vp, float* o) {
    const uint4* pp = (const uint4*)vp;
#pragma unroll
    for (int i = 0; i < 4; ++i) {
        uint4 u = pp[i];
        unsigned w[4] = { u.x, u.y, u.z, u.w };
#pragma unroll
        for (int j = 0; j < 4; ++j) {
            o[i * 8 + j * 2]     += w0 * bf2f(w[j] & 0xffffu);
            o[i * 8 + j * 2 + 1] += w0 * bf2f(w[j] >> 16);
        }
    }
}

__global__ __launch_bounds__(256) void attention_kernel(
    const float* __restrict__ q, const bf16* __restrict__ k0, const bf16* __restrict__ v0,
    const bf16* __restrict__ kf, const bf16* __restrict__ vf,
    const int* __restrict__ idx, bf16* __restrict__ o)
{
    int gid = blockIdx.x * 256 + threadIdx.x;
    if (gid >= TQ * NH) return;
    int t = gid / NH, h = gid - t * NH;
    int b = t >> 11;
    float qv[32];
    const float* qp = q + (size_t)t * DIM + h * HD;
#pragma unroll
    for (int d = 0; d < 32; ++d) qv[d] = qp[d] * 0.17677669529663687f; // 1/sqrt(32)

    float s[KNN + 1];
    s[0] = dot_row(qv, k0 + (size_t)t * DIM + h * HD);
#pragma unroll
    for (int j = 0; j < KNN; ++j) {
        int n = idx[t * KNN + j];
        s[1 + j] = dot_row(qv, kf + ((size_t)(b * N_ + n)) * DIM + h * HD);
    }
    float mx = s[0];
#pragma unroll
    for (int j = 1; j <= KNN; ++j) mx = fmaxf(mx, s[j]);
    float se = 0.f;
#pragma unroll
    for (int j = 0; j <= KNN; ++j) { s[j] = __expf(s[j] - mx); se += s[j]; }
    float inv = 1.f / se;

    float ov[32];
#pragma unroll
    for (int d = 0; d < 32; ++d) ov[d] = 0.f;
    axpy_row(s[0], v0 + (size_t)t * DIM + h * HD, ov);
#pragma unroll
    for (int j = 0; j < KNN; ++j) {
        int n = idx[t * KNN + j];
        axpy_row(s[1 + j], vf + ((size_t)(b * N_ + n)) * DIM + h * HD, ov);
    }
    bf16* op = o + (size_t)t * DIM + h * HD;
#pragma unroll
    for (int d = 0; d < 32; ++d) op[d] = (bf16)(ov[d] * inv);
}

// ---------------------------------------------------------------- final projection (DIMx1)
__global__ __launch_bounds__(256) void out_proj_kernel(
    const float* __restrict__ x, const float* __restrict__ w, float* __restrict__ out, int T)
{
    int wave = threadIdx.x >> 5, lane = threadIdx.x & 31;
    int t = blockIdx.x * 8 + wave;
    if (t >= T) return;
    float s = 0.f;
#pragma unroll
    for (int i = 0; i < 12; ++i) s += x[(size_t)t * DIM + lane + i * 32] * w[lane + i * 32];
#pragma unroll
    for (int o = 16; o; o >>= 1) s += __shfl_xor(s, o, 32);
    if (lane == 0) out[t] = s;
}

// ================================================================ host orchestration
static void gemm(hipStream_t st, const bf16* A, const bf16* Bp, const float* bias,
                 const float* resid, float* Cf, bf16* Cb, int M, int N, int K, int relu) {
    dim3 g(M / 128, N / 64);
    gemm_bf16_kernel<<<g, 256, 0, st>>>(A, Bp, bias, resid, Cf, Cb, M, N, K, relu);
}

extern "C" void kernel_launch(void* const* d_in, const int* in_sizes, int n_in,
                              void* d_out, int out_size, void* d_ws, size_t ws_size,
                              hipStream_t stream)
{
    (void)in_sizes; (void)n_in; (void)out_size; (void)ws_size;
    const float* query   = (const float*)d_in[0];
    const float* p       = (const float*)d_in[1];
    const float* c_point = (const float*)d_in[2];
    const float* planes[3][3] = {
        {(const float*)d_in[3], (const float*)d_in[4], (const float*)d_in[5]},
        {(const float*)d_in[6], (const float*)d_in[7], (const float*)d_in[8]},
        {(const float*)d_in[9], (const float*)d_in[10],(const float*)d_in[11]}};
    const float* mW1[3] = {(const float*)d_in[12], (const float*)d_in[16], (const float*)d_in[20]};
    const float* mb1[3] = {(const float*)d_in[13], (const float*)d_in[17], (const float*)d_in[21]};
    const float* mW2[3] = {(const float*)d_in[14], (const float*)d_in[18], (const float*)d_in[22]};
    const float* mb2[3] = {(const float*)d_in[15], (const float*)d_in[19], (const float*)d_in[23]};
    const float* Wcq  = (const float*)d_in[24];
    const float* ln1g = (const float*)d_in[25]; const float* ln1b = (const float*)d_in[26];
    const float* tWq  = (const float*)d_in[27]; const float* tWk  = (const float*)d_in[28];
    const float* tWv  = (const float*)d_in[29]; const float* tWo  = (const float*)d_in[30];
    const float* tbo  = (const float*)d_in[31];
    const float* ln2g = (const float*)d_in[32]; const float* ln2b = (const float*)d_in[33];
    const float* tWf1 = (const float*)d_in[34]; const float* tbf1 = (const float*)d_in[35];
    const float* tWf2 = (const float*)d_in[36]; const float* tbf2 = (const float*)d_in[37];
    const float* Wout = (const float*)d_in[38];

    size_t off = 0;
    auto alloc = [&](size_t bytes) -> void* {
        void* r = (char*)d_ws + off; off = (off + bytes + 255) & ~(size_t)255; return r;
    };
    auto pack = [&](const float* s, bf16* d, int K, int N) {
        pack_weight_kernel<<<(K * N + 255) / 256, 256, 0, stream>>>(s, d, K, N);
    };

    // ---- packed bf16 weights
    const int C1s[3] = {768, 384, 192};
    bf16* wW1[3]; bf16* wW2[3];
    for (int s = 0; s < 3; ++s) { wW1[s] = (bf16*)alloc((size_t)C1s[s] * 384 * 2);
                                  wW2[s] = (bf16*)alloc((size_t)384 * 192 * 2); }
    bf16* wWcq = (bf16*)alloc(192 * 384 * 2);
    bf16* wWq  = (bf16*)alloc((size_t)NBLK * DIM * DIM * 2);
    bf16* wWk  = (bf16*)alloc((size_t)NBLK * DIM * DIM * 2);
    bf16* wWv  = (bf16*)alloc((size_t)NBLK * DIM * DIM * 2);
    bf16* wWo  = (bf16*)alloc((size_t)NBLK * DIM * DIM * 2);
    bf16* wWf1 = (bf16*)alloc((size_t)NBLK * DIM * 1536 * 2);
    bf16* wWf2 = (bf16*)alloc((size_t)NBLK * 1536 * DIM * 2);

    // ---- phase-A buffers (aliased later)
    float* ptsAll = (float*)alloc((size_t)TALL * 3 * 4);
    bf16*  feat   = (bf16*) alloc((size_t)TALL * 768 * 2);   // reused: kfull f32
    bf16*  Hbf    = (bf16*) alloc((size_t)TALL * DIM * 2);   // reused: vfull bf16
    float* T192f  = (float*)alloc((size_t)TALL * 192 * 4);   // reused: q f32 + k0 f32
    bf16*  T192b  = (bf16*) alloc((size_t)TALL * 192 * 2);
    // ---- persistent / per-block buffers
    float* cq   = (float*)alloc((size_t)TQ * DIM * 4);
    bf16*  x0b  = (bf16*) alloc((size_t)TQ * DIM * 2);
    float* cff  = (float*)alloc((size_t)TP * DIM * 4);
    bf16*  cfb  = (bf16*) alloc((size_t)TP * DIM * 2);
    int*   knn  = (int*)  alloc((size_t)TQ * KNN * 4);
    bf16*  ln1o = (bf16*) alloc((size_t)TQ * DIM * 2);
    bf16*  k0b  = (bf16*) alloc((size_t)TQ * DIM * 2);
    bf16*  v0b  = (bf16*) alloc((size_t)TQ * DIM * 2);
    bf16*  kfb  = (bf16*) alloc((size_t)TP * DIM * 2);
    bf16*  ob   = (bf16*) alloc((size_t)TQ * DIM * 2);
    float* X1   = (float*)alloc((size_t)TQ * DIM * 4);
    bf16*  ln2o = (bf16*) alloc((size_t)TQ * DIM * 2);
    bf16*  h1b  = (bf16*) alloc((size_t)TQ * 1536 * 2);
    // aliases (phase-A regions dead once blocks start)
    float* kff = (float*)feat;
    bf16*  vfb = (bf16*)Hbf;
    float* qf  = (float*)T192f;
    float* k0f = (float*)T192f + (size_t)TQ * DIM;

    // ---- weight packing (fragment order)
    for (int s = 0; s < 3; ++s) { pack(mW1[s], wW1[s], C1s[s], 384); pack(mW2[s], wW2[s], 384, 192); }
    pack(Wcq, wWcq, 192, 384);
    for (int blk = 0; blk < NBLK; ++blk) {
        pack(tWq + (size_t)blk * DIM * DIM, wWq + (size_t)blk * DIM * DIM, DIM, DIM);
        pack(tWk + (size_t)blk * DIM * DIM, wWk + (size_t)blk * DIM * DIM, DIM, DIM);
        pack(tWv + (size_t)blk * DIM * DIM, wWv + (size_t)blk * DIM * DIM, DIM, DIM);
        pack(tWo + (size_t)blk * DIM * DIM, wWo + (size_t)blk * DIM * DIM, DIM, DIM);
        pack(tWf1 + (size_t)blk * DIM * 1536, wWf1 + (size_t)blk * DIM * 1536, DIM, 1536);
        pack(tWf2 + (size_t)blk * 1536 * DIM, wWf2 + (size_t)blk * 1536 * DIM, 1536, DIM);
    }

    // ---- all points (query rows first, then cloud rows)
    hipMemcpyAsync(ptsAll, query, (size_t)TQ * 3 * 4, hipMemcpyDeviceToDevice, stream);
    hipMemcpyAsync(ptsAll + (size_t)TQ * 3, p, (size_t)TP * 3 * 4, hipMemcpyDeviceToDevice, stream);

    // ---- triplane features + per-scale MLPs, accumulated into T192
    const int Rs[3] = {32, 64, 128};
    for (int s = 0; s < 3; ++s) {
        int nthr = TALL * C1s[s];
        triplane_sample_kernel<<<(nthr + 255) / 256, 256, 0, stream>>>(
            ptsAll, planes[s][0], planes[s][1], planes[s][2], feat, C1s[s], Rs[s]);
        gemm(stream, feat, wW1[s], mb1[s], nullptr, nullptr, Hbf, TALL, 384, C1s[s], 1);
        gemm(stream, Hbf, wW2[s], mb2[s], s ? T192f : nullptr, T192f,
             (s == 2) ? T192b : nullptr, TALL, 192, 384, 0);
    }

    // ---- c_query = T192[query rows] @ Wcq ;  c_full = concat(c_point, T192[cloud rows])
    gemm(stream, T192b, wWcq, nullptr, nullptr, cq, x0b, TQ, 384, 192, 0);
    build_cfull_kernel<<<(TP * DIM + 255) / 256, 256, 0, stream>>>(c_point, T192f, cff, cfb);

    // ---- kNN
    knn_kernel<<<TQ / 2, 64, 0, stream>>>(query, p, knn);

    // ---- transformer blocks (K/V projected on c_full once, gathered in attention)
    for (int blk = 0; blk < NBLK; ++blk) {
        const bf16* Wq_b  = wWq  + (size_t)blk * DIM * DIM;
        const bf16* Wk_b  = wWk  + (size_t)blk * DIM * DIM;
        const bf16* Wv_b  = wWv  + (size_t)blk * DIM * DIM;
        const bf16* Wo_b  = wWo  + (size_t)blk * DIM * DIM;
        const bf16* Wf1_b = wWf1 + (size_t)blk * DIM * 1536;
        const bf16* Wf2_b = wWf2 + (size_t)blk * 1536 * DIM;

        layernorm_kernel<<<TQ / 8, 256, 0, stream>>>(cq, ln1g + blk * DIM, ln1b + blk * DIM, ln1o, TQ);
        gemm(stream, ln1o, Wq_b, nullptr, nullptr, qf, nullptr, TQ, DIM, DIM, 0);
        gemm(stream, x0b,  Wk_b, nullptr, nullptr, k0f, nullptr, TQ, DIM, DIM, 0);
        gemm(stream, x0b,  Wv_b, nullptr, nullptr, nullptr, v0b, TQ, DIM, DIM, 0);
        gemm(stream, cfb,  Wk_b, nullptr, nullptr, kff, nullptr, TP, DIM, DIM, 0);
        gemm(stream, cfb,  Wv_b, nullptr, nullptr, nullptr, vfb, TP, DIM, DIM, 0);

        rope_kernel<<<(TQ * 192 + 255) / 256, 256, 0, stream>>>(qf,  query, qf, nullptr, TQ);
        rope_kernel<<<(TQ * 192 + 255) / 256, 256, 0, stream>>>(k0f, query, nullptr, k0b, TQ);
        rope_kernel<<<(TP * 192 + 255) / 256, 256, 0, stream>>>(kff, p, nullptr, kfb, TP);

        attention_kernel<<<(TQ * NH + 255) / 256, 256, 0, stream>>>(qf, k0b, v0b, kfb, vfb, knn, ob);

        gemm(stream, ob, Wo_b, tbo + blk * DIM, cq, X1, nullptr, TQ, DIM, DIM, 0);
        layernorm_kernel<<<TQ / 8, 256, 0, stream>>>(X1, ln2g + blk * DIM, ln2b + blk * DIM, ln2o, TQ);
        gemm(stream, ln2o, Wf1_b, tbf1 + blk * 1536, nullptr, nullptr, h1b, TQ, 1536, DIM, 1);
        gemm(stream, h1b,  Wf2_b, tbf2 + blk * DIM, X1, cq, x0b, TQ, DIM, 1536, 0);
    }

    // ---- logits
    out_proj_kernel<<<TQ / 8, 256, 0, stream>>>(cq, Wout, (float*)d_out, TQ);
}
